// GlobalContextModule_81819126989315
// MI455X (gfx1250) — matrix operand order repaired and checked by reading, not compiled, not run on hardware
//
#include <hip/hip_runtime.h>
#include <hip/hip_bf16.h>
#include <stdint.h>

// ---------------------------------------------------------------------------
// Sizes (fixed by the reference): B=8, C=256, N=64*64=4096, inter=32
// ---------------------------------------------------------------------------
#define BB 8
#define CC 256
#define NN 4096
#define II 32

typedef __attribute__((ext_vector_type(16))) __bf16 v16bf;
typedef __attribute__((ext_vector_type(8)))  float  v8f;

union FragBF {
    v16bf    f;
    uint4    u[2];
    uint32_t w[8];
};

union Pack8 {            // 8 bf16 = 16 bytes (one b128 LDS/VMEM access)
    uint4    u4;
    uint16_t h[8];
    uint32_t w[4];
};

// native bf16 converts (compiler emits v_cvt_* ops)
__device__ __forceinline__ uint16_t f2bf(float a) {
    __bf16 h = (__bf16)a;
    return __builtin_bit_cast(uint16_t, h);
}
__device__ __forceinline__ float bf2f(uint16_t u) {
    return (float)__builtin_bit_cast(__bf16, u);
}

// ---------------------------------------------------------------------------
// Kernel 0: one-shot weight convert f32 -> bf16 (row-major, tiny)
// ---------------------------------------------------------------------------
__global__ __launch_bounds__(256) void k_wconv(const float* __restrict__ Wq,
                                               const float* __restrict__ Wk,
                                               const float* __restrict__ Wv,
                                               uint16_t* __restrict__ Wqb,
                                               uint16_t* __restrict__ Wkb,
                                               uint16_t* __restrict__ Wvb) {
    int i = blockIdx.x * 256 + threadIdx.x;          // grid covers 65536
    if (i < II * CC) {
        Wqb[i] = f2bf(Wq[i]);
        Wkb[i] = f2bf(Wk[i]);
    }
    if (i < CC * CC) Wvb[i] = f2bf(Wv[i]);
}

// ---------------------------------------------------------------------------
// Kernel 1: x [B,C,N] f32  ->  xT [B,N,C] bf16   (LDS 32x32 tile transpose)
// ---------------------------------------------------------------------------
__global__ __launch_bounds__(256) void k_transpose(const float* __restrict__ x,
                                                   uint16_t* __restrict__ xT) {
    __shared__ float tile[32][33];
    int bid  = blockIdx.x;
    int nt   = bid & (NN / 32 - 1);
    int rest = bid >> 7;
    int ct   = rest & (CC / 32 - 1);
    int b    = rest >> 3;
    int tid  = threadIdx.x;
    int tn = tid & 31, tc = tid >> 5;
    const float* xp = x + ((size_t)(b * CC + ct * 32)) * NN + nt * 32;
#pragma unroll
    for (int i = 0; i < 32; i += 8)
        tile[tc + i][tn] = xp[(size_t)(tc + i) * NN + tn];
    __syncthreads();
    int wc = tid & 31, wn = tid >> 5;
    uint16_t* op = xT + ((size_t)(b * NN + nt * 32)) * CC + ct * 32;
#pragma unroll
    for (int i = 0; i < 32; i += 8)
        op[(size_t)(wn + i) * CC + wc] = f2bf(tile[wc][wn + i]);
}

// ---------------------------------------------------------------------------
// Kernel 2: global average pool  gp[b,c] = mean_n x[b,c,n]
// ---------------------------------------------------------------------------
__global__ __launch_bounds__(256) void k_gap(const float* __restrict__ x,
                                             float* __restrict__ gp) {
    int bc = blockIdx.x;
    const float* xp = x + (size_t)bc * NN;
    float s = 0.f;
    for (int i = threadIdx.x; i < NN; i += 256) s += xp[i];
    __shared__ float red[256];
    red[threadIdx.x] = s;
    __syncthreads();
#pragma unroll
    for (int off = 128; off > 0; off >>= 1) {
        if (threadIdx.x < off) red[threadIdx.x] += red[threadIdx.x + off];
        __syncthreads();
    }
    if (threadIdx.x == 0) gp[bc] = red[0] * (1.0f / NN);
}

// ---------------------------------------------------------------------------
// Kernel 3: gate MLP:  ga1p[b,c] = 1 + sigmoid(Wg2 · relu(Wg1·gp + bg1) + bg2)
// ---------------------------------------------------------------------------
__global__ __launch_bounds__(256) void k_gate(const float* __restrict__ gp,
                                              const float* __restrict__ Wg1,
                                              const float* __restrict__ bg1,
                                              const float* __restrict__ Wg2,
                                              const float* __restrict__ bg2,
                                              float* __restrict__ ga1p) {
    int b = blockIdx.x;
    __shared__ float sgp[CC];
    __shared__ float sh[II];
    sgp[threadIdx.x] = gp[b * CC + threadIdx.x];
    __syncthreads();
    if (threadIdx.x < II) {
        float a = bg1[threadIdx.x];
        const float* wr = Wg1 + threadIdx.x * CC;
        for (int c = 0; c < CC; ++c) a += wr[c] * sgp[c];
        sh[threadIdx.x] = fmaxf(a, 0.f);
    }
    __syncthreads();
    float a = bg2[threadIdx.x];
    const float* wr = Wg2 + threadIdx.x * II;
#pragma unroll
    for (int o = 0; o < II; ++o) a += wr[o] * sh[o];
    float sg = 1.0f / (1.0f + __builtin_amdgcn_exp2f(-a * 1.44269504f));
    ga1p[b * CC + threadIdx.x] = 1.0f + sg;
}

// ---------------------------------------------------------------------------
// Kernel 4: q/k projection (WMMA).  D[n,o] = xT(16n x 32c) * W^T(32c x 16o)
// ---------------------------------------------------------------------------
__global__ __launch_bounds__(256) void k_proj_qk(const uint16_t* __restrict__ xT,
                                                 const uint16_t* __restrict__ Wqb,
                                                 const float* __restrict__ bq,
                                                 const uint16_t* __restrict__ Wkb,
                                                 const float* __restrict__ bk,
                                                 uint16_t* __restrict__ qT,
                                                 uint16_t* __restrict__ kT) {
    int wave  = blockIdx.x * 8 + (threadIdx.x >> 5);
    int lane  = threadIdx.x & 31;
    int hf    = lane >> 4, l16 = lane & 15;
    int which = wave & 1;
    int t     = wave >> 1;
    int ntile = t & (NN / 16 - 1);
    int b     = t >> 8;
    const uint16_t* W    = which ? Wkb : Wqb;
    const float*    bias = which ? bk : bq;
    uint16_t*       out  = which ? kT : qT;
    int n0 = ntile * 16;

    const uint16_t* xbase = xT + ((size_t)(b * NN + n0 + l16)) * CC;
    const uint16_t* w0 = W + (size_t)l16 * CC + hf * 8;
    const uint16_t* w1 = W + (size_t)(16 + l16) * CC + hf * 8;
    v8f acc0 = {}, acc1 = {};
    for (int kc = 0; kc < CC; kc += 32) {
        FragBF A;
        A.u[0] = *(const uint4*)(xbase + kc + hf * 8);
        A.u[1] = *(const uint4*)(xbase + kc + 16 + hf * 8);
        FragBF B0, B1;
        B0.u[0] = *(const uint4*)(w0 + kc);
        B0.u[1] = *(const uint4*)(w0 + kc + 16);
        B1.u[0] = *(const uint4*)(w1 + kc);
        B1.u[1] = *(const uint4*)(w1 + kc + 16);
        acc0 = __builtin_amdgcn_wmma_f32_16x16x32_bf16(false, A.f, false, B0.f,
                                                       (short)0, acc0, false, false);
        acc1 = __builtin_amdgcn_wmma_f32_16x16x32_bf16(false, A.f, false, B1.f,
                                                       (short)0, acc1, false, false);
    }
    uint16_t* op = out + (size_t)b * NN * II;
    float bo0 = bias[l16], bo1 = bias[16 + l16];
#pragma unroll
    for (int v = 0; v < 8; ++v) {
        int n = n0 + v + 8 * hf;
        op[(size_t)n * II + l16]      = f2bf(acc0[v] + bo0);
        op[(size_t)n * II + 16 + l16] = f2bf(acc1[v] + bo1);
    }
}

// ---------------------------------------------------------------------------
// Kernel 5: v projection (WMMA). D[c,n] = Wv(16c x 32k) * xT^T(32k x 16n)
// ---------------------------------------------------------------------------
__global__ __launch_bounds__(256) void k_proj_v(const uint16_t* __restrict__ xT,
                                                const uint16_t* __restrict__ Wvb,
                                                const float* __restrict__ bv,
                                                uint16_t* __restrict__ vout) {
    int wave  = blockIdx.x * 8 + (threadIdx.x >> 5);
    int lane  = threadIdx.x & 31;
    int hf    = lane >> 4, l16 = lane & 15;
    int ntile = wave & (NN / 16 - 1);
    int rest  = wave >> 8;
    int ct    = rest & 15;
    int b     = rest >> 4;
    int c0 = ct * 16, n0 = ntile * 16;

    v8f acc = {};
    const uint16_t* wrow = Wvb + (size_t)(c0 + l16) * CC;
    const uint16_t* xb0  = xT + ((size_t)(b * NN + n0 + l16)) * CC + hf * 8;
    for (int kc = 0; kc < CC; kc += 32) {
        FragBF A;
        A.u[0] = *(const uint4*)(wrow + kc + hf * 8);
        A.u[1] = *(const uint4*)(wrow + kc + 16 + hf * 8);
        FragBF Bf;
        Bf.u[0] = *(const uint4*)(xb0 + kc);
        Bf.u[1] = *(const uint4*)(xb0 + kc + 16);
        acc = __builtin_amdgcn_wmma_f32_16x16x32_bf16(false, A.f, false, Bf.f,
                                                      (short)0, acc, false, false);
    }
#pragma unroll
    for (int v = 0; v < 8; ++v) {
        int c = c0 + v + 8 * hf;
        vout[((size_t)(b * CC + c)) * NN + n0 + l16] = f2bf(acc[v] + bv[c]);
    }
}

// ---------------------------------------------------------------------------
// Kernel 6: fused attention, 32 n-rows per workgroup (two 16-row strips).
//  Phase A: S^T[m,n] = K^T Q ; each K fragment feeds 2 Q fragments (2 WMMA)
//  Phase B: per-column max -> exp2 -> sum, in place, b128 LDS traffic
//  Phase C: out^T[c,n] = V * P^T ; each V fragment feeds 2 P fragments
//           -> 4 WMMAs per 8 b128 loads; waves split the 256 channels.
// LDS: 256KB logits (CDNA5: 320KB/WGP) + reductions + gate cache.
// ---------------------------------------------------------------------------
__global__ __launch_bounds__(256, 1) void k_attn(const uint16_t* __restrict__ qT,
                                                 const uint16_t* __restrict__ kT,
                                                 const uint16_t* __restrict__ vmat,
                                                 const float* __restrict__ x,
                                                 const float* __restrict__ gamma,
                                                 const float* __restrict__ ga1p,
                                                 float* __restrict__ out) {
    extern __shared__ char smem[];
    uint16_t* S    = (uint16_t*)smem;                    // [32][4096] bf16
    float*    red  = (float*)(smem + 32 * NN * 2);       // [8][32]
    float*    sga  = red + 256;                          // [256]

    int tid = threadIdx.x;
    int w = tid >> 5, lane = tid & 31, hf = lane >> 4, l16 = lane & 15;
    int b     = blockIdx.x >> 7;                 // N/32 = 128 tiles per batch
    int ntile = blockIdx.x & (NN / 32 - 1);
    int n0    = ntile * 32;                      // strip0: n0.., strip1: n0+16..

    sga[tid] = ga1p[b * CC + tid];

    // Q B-fragments for the two strips
    FragBF Qf0, Qf1;
    {
        const uint16_t* qb0 = qT + ((size_t)(b * NN + n0 + l16)) * II + hf * 8;
        const uint16_t* qb1 = qb0 + 16 * II;
        Qf0.u[0] = *(const uint4*)qb0;
        Qf0.u[1] = *(const uint4*)(qb0 + 16);
        Qf1.u[0] = *(const uint4*)qb1;
        Qf1.u[1] = *(const uint4*)(qb1 + 16);
    }

    // ---- Phase A: logits, one K fragment -> two WMMAs --------------------
    int mbase = w * (NN / 8);                    // 512 m per wave
    float mymax0 = -3.0e38f, mymax1 = -3.0e38f;
    for (int mc = 0; mc < NN / 8; mc += 16) {
        int m0 = mbase + mc;
        const uint16_t* kb = kT + ((size_t)(b * NN + m0 + l16)) * II;
        FragBF A;
        A.u[0] = *(const uint4*)(kb + hf * 8);
        A.u[1] = *(const uint4*)(kb + 16 + hf * 8);
        v8f d0 = {}, d1 = {};
        d0 = __builtin_amdgcn_wmma_f32_16x16x32_bf16(false, A.f, false, Qf0.f,
                                                     (short)0, d0, false, false);
        d1 = __builtin_amdgcn_wmma_f32_16x16x32_bf16(false, A.f, false, Qf1.f,
                                                     (short)0, d1, false, false);
        Pack8 p0, p1;   // lane holds 8 contiguous m: m0 + 8*hf + 0..7
#pragma unroll
        for (int v = 0; v < 8; ++v) {
            float s0 = d0[v], s1 = d1[v];
            mymax0 = fmaxf(mymax0, s0);
            mymax1 = fmaxf(mymax1, s1);
            p0.h[v] = f2bf(s0);
            p1.h[v] = f2bf(s1);
        }
        *(uint4*)(&S[l16 * NN + m0 + 8 * hf])        = p0.u4;
        *(uint4*)(&S[(16 + l16) * NN + m0 + 8 * hf]) = p1.u4;
    }
    mymax0 = fmaxf(mymax0, __shfl_xor(mymax0, 16, 32));
    mymax1 = fmaxf(mymax1, __shfl_xor(mymax1, 16, 32));
    if (lane < 16) {
        red[w * 32 + l16]      = mymax0;
        red[w * 32 + 16 + l16] = mymax1;
    }
    __syncthreads();
    float gmax0 = -3.0e38f, gmax1 = -3.0e38f;
#pragma unroll
    for (int i = 0; i < 8; ++i) {
        gmax0 = fmaxf(gmax0, red[i * 32 + l16]);
        gmax1 = fmaxf(gmax1, red[i * 32 + 16 + l16]);
    }
    __syncthreads();   // reads of red(max) done before reuse

    // ---- Phase B: exp + column sums, in place ----------------------------
    float mysum0 = 0.f, mysum1 = 0.f;
    for (int mc = 0; mc < NN / 8; mc += 16) {
        int m0 = mbase + mc;
        uint4* sp0 = (uint4*)(&S[l16 * NN + m0 + 8 * hf]);
        uint4* sp1 = (uint4*)(&S[(16 + l16) * NN + m0 + 8 * hf]);
        Pack8 p0, p1;
        p0.u4 = *sp0;
        p1.u4 = *sp1;
#pragma unroll
        for (int v = 0; v < 8; ++v) {
            float e0 = __builtin_amdgcn_exp2f((bf2f(p0.h[v]) - gmax0) * 1.44269504f);
            float e1 = __builtin_amdgcn_exp2f((bf2f(p1.h[v]) - gmax1) * 1.44269504f);
            mysum0 += e0;
            mysum1 += e1;
            p0.h[v] = f2bf(e0);
            p1.h[v] = f2bf(e1);
        }
        *sp0 = p0.u4;
        *sp1 = p1.u4;
    }
    mysum0 += __shfl_xor(mysum0, 16, 32);
    mysum1 += __shfl_xor(mysum1, 16, 32);
    if (lane < 16) {
        red[w * 32 + l16]      = mysum0;
        red[w * 32 + 16 + l16] = mysum1;
    }
    __syncthreads();   // also publishes all p writes to S
    float gsum0 = 0.f, gsum1 = 0.f;
#pragma unroll
    for (int i = 0; i < 8; ++i) {
        gsum0 += red[i * 32 + l16];
        gsum1 += red[i * 32 + 16 + l16];
    }
    float rsum0 = 1.0f / gsum0, rsum1 = 1.0f / gsum1;

    // ---- Phase C: PV. wave handles channels [w*32, w*32+32) --------------
    int c0 = w * 32;
    v8f a00 = {}, a01 = {}, a10 = {}, a11 = {};
    const uint16_t* vb0 = vmat + ((size_t)(b * CC + c0 + l16)) * NN;
    const uint16_t* vb1 = vb0 + (size_t)16 * NN;
    const uint16_t* Sl0 = S + l16 * NN + hf * 8;
    const uint16_t* Sl1 = S + (16 + l16) * NN + hf * 8;
    for (int m0 = 0; m0 < NN; m0 += 32) {
        __builtin_prefetch(vb0 + m0 + 2048, 0, 0);   // ~4KB ahead of the V stream
        __builtin_prefetch(vb1 + m0 + 2048, 0, 0);
        FragBF P0, P1;
        P0.u[0] = *(const uint4*)(Sl0 + m0);
        P0.u[1] = *(const uint4*)(Sl0 + m0 + 16);
        P1.u[0] = *(const uint4*)(Sl1 + m0);
        P1.u[1] = *(const uint4*)(Sl1 + m0 + 16);
        FragBF A0, A1;
        A0.u[0] = *(const uint4*)(vb0 + m0 + hf * 8);
        A0.u[1] = *(const uint4*)(vb0 + m0 + 16 + hf * 8);
        A1.u[0] = *(const uint4*)(vb1 + m0 + hf * 8);
        A1.u[1] = *(const uint4*)(vb1 + m0 + 16 + hf * 8);
        a00 = __builtin_amdgcn_wmma_f32_16x16x32_bf16(false, A0.f, false, P0.f,
                                                      (short)0, a00, false, false);
        a01 = __builtin_amdgcn_wmma_f32_16x16x32_bf16(false, A0.f, false, P1.f,
                                                      (short)0, a01, false, false);
        a10 = __builtin_amdgcn_wmma_f32_16x16x32_bf16(false, A1.f, false, P0.f,
                                                      (short)0, a10, false, false);
        a11 = __builtin_amdgcn_wmma_f32_16x16x32_bf16(false, A1.f, false, P1.f,
                                                      (short)0, a11, false, false);
    }

    // ---- Epilogue: 1/sum, gamma*out + residual, * (1 + gate) -------------
    float g = gamma[0];
    const float* xb = x + (size_t)b * CC * NN;
    float*       ob = out + (size_t)b * CC * NN;
    int na = n0 + l16, nb = n0 + 16 + l16;
#pragma unroll
    for (int v = 0; v < 8; ++v) {
        int c = c0 + v + 8 * hf;
        float ga = sga[c];
        float o0 = g * (a00[v] * rsum0) + xb[(size_t)c * NN + na];
        float o1 = g * (a01[v] * rsum1) + xb[(size_t)c * NN + nb];
        ob[(size_t)c * NN + na] = o0 * ga;
        ob[(size_t)c * NN + nb] = o1 * ga;
    }
#pragma unroll
    for (int v = 0; v < 8; ++v) {
        int c = c0 + 16 + v + 8 * hf;
        float ga = sga[c];
        float o0 = g * (a10[v] * rsum0) + xb[(size_t)c * NN + na];
        float o1 = g * (a11[v] * rsum1) + xb[(size_t)c * NN + nb];
        ob[(size_t)c * NN + na] = o0 * ga;
        ob[(size_t)c * NN + nb] = o1 * ga;
    }
}

// ---------------------------------------------------------------------------
// Host launcher
// ---------------------------------------------------------------------------
extern "C" void kernel_launch(void* const* d_in, const int* in_sizes, int n_in,
                              void* d_out, int out_size, void* d_ws, size_t ws_size,
                              hipStream_t stream) {
    const float* x     = (const float*)d_in[0];
    const float* Wq    = (const float*)d_in[1];
    const float* bq    = (const float*)d_in[2];
    const float* Wk    = (const float*)d_in[3];
    const float* bk    = (const float*)d_in[4];
    const float* Wv    = (const float*)d_in[5];
    const float* bv    = (const float*)d_in[6];
    const float* gamma = (const float*)d_in[7];
    const float* Wg1   = (const float*)d_in[8];
    const float* bg1   = (const float*)d_in[9];
    const float* Wg2   = (const float*)d_in[10];
    const float* bg2   = (const float*)d_in[11];
    float* out = (float*)d_out;

    char* ws = (char*)d_ws;
    size_t off = 0;
    uint16_t* xT = (uint16_t*)(ws + off); off += (size_t)BB * NN * CC * 2;   // 16 MB
    uint16_t* qT = (uint16_t*)(ws + off); off += (size_t)BB * NN * II * 2;   //  2 MB
    uint16_t* kT = (uint16_t*)(ws + off); off += (size_t)BB * NN * II * 2;   //  2 MB
    uint16_t* vw = (uint16_t*)(ws + off); off += (size_t)BB * CC * NN * 2;   // 16 MB
    float*    gp   = (float*)(ws + off);  off += (size_t)BB * CC * 4;
    float*    ga1p = (float*)(ws + off);  off += (size_t)BB * CC * 4;
    uint16_t* Wqb = (uint16_t*)(ws + off); off += (size_t)II * CC * 2;
    uint16_t* Wkb = (uint16_t*)(ws + off); off += (size_t)II * CC * 2;
    uint16_t* Wvb = (uint16_t*)(ws + off); off += (size_t)CC * CC * 2;

    // 0. weight convert (one shot, tiny)
    k_wconv<<<(CC * CC) / 256, 256, 0, stream>>>(Wq, Wk, Wv, Wqb, Wkb, Wvb);
    // 1. transpose + bf16 convert
    k_transpose<<<BB * (CC / 32) * (NN / 32), 256, 0, stream>>>(x, xT);
    // 2-3. gating branch (tiny)
    k_gap<<<BB * CC, 256, 0, stream>>>(x, gp);
    k_gate<<<BB, 256, 0, stream>>>(gp, Wg1, bg1, Wg2, bg2, ga1p);
    // 4-5. WMMA projections
    k_proj_qk<<<(BB * (NN / 16) * 2) / 8, 256, 0, stream>>>(xT, Wqb, bq, Wkb, bk, qT, kT);
    k_proj_v<<<(BB * (CC / 16) * (NN / 16)) / 8, 256, 0, stream>>>(xT, Wvb, bv, vw);
    // 6. fused flash attention + epilogue (32 n-rows / workgroup, 258KB LDS)
    const int smem = 32 * NN * 2 + (256 + 256) * 4;   // 264192 B
    static bool attr_set = [] {
        (void)hipFuncSetAttribute((const void*)k_attn,
                                  hipFuncAttributeMaxDynamicSharedMemorySize,
                                  32 * NN * 2 + (256 + 256) * 4);
        return true;
    }();
    (void)attr_set;
    k_attn<<<BB * (NN / 32), 256, smem, stream>>>(qT, kT, vw, x, gamma, ga1p, out);
}
